// EncoderBlock_52458730553452
// MI455X (gfx1250) — matrix operand order repaired and checked by reading, not compiled, not run on hardware
//
#include <hip/hip_runtime.h>
#include <hip/hip_bf16.h>
#include <stdint.h>

// ---------------- problem constants ----------------
static constexpr int Bb  = 4;
static constexpr int S   = 2048;
static constexpr int D   = 1024;
static constexpr int H   = 16;
static constexpr int HD  = 64;
static constexpr int DFF = 4096;
static constexpr int M   = Bb * S;          // 8192 rows
static constexpr float EPS = 1e-6f;

typedef _Float16 h16 __attribute__((ext_vector_type(16)));
typedef _Float16 h8v __attribute__((ext_vector_type(8)));
typedef float    f8  __attribute__((ext_vector_type(8)));
typedef int      v4i __attribute__((vector_size(16)));   // matches builtin param type

// -------- CDNA5 async global->LDS copy (ASYNCcnt-tracked) --------
#if defined(__has_builtin)
#if __has_builtin(__builtin_amdgcn_global_load_async_to_lds_b128)
#define HAVE_ASYNC_LDS 1
#else
#define HAVE_ASYNC_LDS 0
#warning "CDNA5 probe: __builtin_amdgcn_global_load_async_to_lds_b128 NOT available - using sync fallback"
#endif
#if !__has_builtin(__builtin_amdgcn_s_wait_asynccnt)
#warning "CDNA5 probe: __builtin_amdgcn_s_wait_asynccnt NOT available"
#endif
#else
#define HAVE_ASYNC_LDS 0
#endif

__device__ __forceinline__ void async_b128(const _Float16* g, _Float16* l) {
#if HAVE_ASYNC_LDS
    __builtin_amdgcn_global_load_async_to_lds_b128(
        (__attribute__((address_space(1))) v4i*)g,
        (__attribute__((address_space(3))) v4i*)l, 0, 0);
#else
    *(h8v*)l = *(const h8v*)g;
#endif
}

__device__ __forceinline__ void wait_async_all() {
#if HAVE_ASYNC_LDS
#if __has_builtin(__builtin_amdgcn_s_wait_asynccnt)
    __builtin_amdgcn_s_wait_asynccnt(0);
#else
    asm volatile("s_wait_asynccnt 0x0" ::: "memory");
#endif
#endif
}

// ds_swizzle with compile-time pattern (Sema requires a constant integer)
template <int IMM>
__device__ __forceinline__ float swz(float v) {
    return __int_as_float(__builtin_amdgcn_ds_swizzle(__float_as_int(v), IMM));
}
// 16-lane butterfly reductions (xor 1/2/4/8, and-mask 0x1f keeps lane-group on wave32)
__device__ __forceinline__ float red_max16(float v) {
    v = fmaxf(v, swz<0x041f>(v));
    v = fmaxf(v, swz<0x081f>(v));
    v = fmaxf(v, swz<0x101f>(v));
    v = fmaxf(v, swz<0x201f>(v));
    return v;
}
__device__ __forceinline__ float red_sum16(float v) {
    v += swz<0x041f>(v);
    v += swz<0x081f>(v);
    v += swz<0x101f>(v);
    v += swz<0x201f>(v);
    return v;
}

// Load a 16-half fragment from two contiguous 8-half (16B) chunks.
__device__ __forceinline__ h16 frag_from2(const _Float16* p0, const _Float16* p1) {
    h8v lo = *(const h8v*)p0;
    h8v hi = *(const h8v*)p1;
    h16 r;
#pragma unroll
    for (int i = 0; i < 8; ++i) { r[i] = lo[i]; r[i + 8] = hi[i]; }
    return r;
}

// ---------------- weight repack: f32 (KxN row-major) -> f16 tile-major [n0/128][k0/32][n128][k32] ----------------
__global__ void k_pack_wt(const float* __restrict__ w, _Float16* __restrict__ wp, int K, int N) {
    const int ktiles = K >> 5;
    const size_t total = (size_t)K * N;
    for (size_t o = (size_t)blockIdx.x * blockDim.x + threadIdx.x; o < total;
         o += (size_t)gridDim.x * blockDim.x) {
        int t  = (int)(o >> 12);          // 4096 halves per 128x32 tile
        int ti = t / ktiles, tj = t % ktiles;
        int nl = (int)(o >> 5) & 127, kl = (int)o & 31;
        wp[o] = (_Float16)w[(size_t)(tj * 32 + kl) * N + ti * 128 + nl];
    }
}

// ---------------- LayerNorm (unbiased std, /(std+eps)), f16 out ----------------
__global__ void k_layernorm(const float* __restrict__ x, const float* __restrict__ alpha,
                            const float* __restrict__ beta, _Float16* __restrict__ out) {
    const int row = blockIdx.x;
    const int tid = threadIdx.x;
    __shared__ float s1[256], s2[256];
    const float* xr = x + (size_t)row * D;
    float sum = 0.f, sq = 0.f;
#pragma unroll
    for (int j = 0; j < 4; ++j) {
        float t = xr[tid + j * 256];
        sum += t; sq += t * t;
    }
    s1[tid] = sum; s2[tid] = sq;
    __syncthreads();
    for (int off = 128; off > 0; off >>= 1) {
        if (tid < off) { s1[tid] += s1[tid + off]; s2[tid] += s2[tid + off]; }
        __syncthreads();
    }
    const float mean = s1[0] * (1.0f / D);
    const float var  = (s2[0] - s1[0] * mean) * (1.0f / (D - 1));
    const float inv  = 1.0f / (sqrtf(var) + EPS);
    _Float16* orow = out + (size_t)row * D;
#pragma unroll
    for (int j = 0; j < 4; ++j) {
        int i = tid + j * 256;
        float t = xr[i];
        orow[i] = (_Float16)(alpha[i] * (t - mean) * inv + beta[i]);
    }
}

// ---------------- WMMA GEMM: C = A(f16,MxK row-major) @ Bp(f16 packed) + bias (+resid)(relu) ----
// OUTMODE: 0 = f32 row-major, 1 = f16 row-major, 2 = f16 V-transposed [b,h][hd][s]
// 256 threads = 8 waves; block tile 128x128; wave tile 32x64; K step 32; double-buffered async LDS.
template <bool RELU, bool RESID, int OUTMODE>
__global__ __launch_bounds__(256)
void k_gemm_wmma(const _Float16* __restrict__ A, const _Float16* __restrict__ Bp,
                 const float* __restrict__ bias, const float* __restrict__ resid,
                 float* __restrict__ C32, _Float16* __restrict__ C16,
                 int N, int K) {
    __shared__ _Float16 lA[2][128 * 40];    // [row][k] stride 40
    __shared__ _Float16 lBt[2][128 * 40];   // [n][k]   stride 40

    const int tid = threadIdx.x;
    const int l   = tid & 31;
    const int w   = tid >> 5;
    const int wm  = w >> 1, wn = w & 1;
    const int g   = l >> 4, ln = l & 15;
    const int m0  = blockIdx.y * 128;
    const int n0  = blockIdx.x * 128;
    const int ktiles = K >> 5;
    const _Float16* BpBase = Bp + (size_t)(n0 >> 7) * ktiles * 4096;

    auto issue = [&](int kt, int buf) {
#pragma unroll
        for (int j = 0; j < 2; ++j) {          // A tile: 128 rows x 32 k
            int chunk = tid + 256 * j;
            int row = chunk >> 2, co = (chunk & 3) * 8;
            async_b128(&A[(size_t)(m0 + row) * K + (kt << 5) + co], &lA[buf][row * 40 + co]);
        }
        const _Float16* bt = BpBase + (size_t)kt * 4096;   // packed [n128][k32]
#pragma unroll
        for (int j = 0; j < 2; ++j) {
            int chunk = tid + 256 * j;
            int nl = chunk >> 2, co = (chunk & 3) * 8;
            async_b128(&bt[nl * 32 + co], &lBt[buf][nl * 40 + co]);
        }
    };

    f8 acc[2][4];
#pragma unroll
    for (int a = 0; a < 2; ++a)
#pragma unroll
        for (int b = 0; b < 4; ++b)
#pragma unroll
            for (int r = 0; r < 8; ++r) acc[a][b][r] = 0.f;

    issue(0, 0);
    for (int it = 0; it < ktiles; ++it) {
        const int buf = it & 1;
        wait_async_all();        // this wave's tile-it copies have landed
        __syncthreads();         // everyone's landed; everyone done reading buf from tile it-1
        if (it + 1 < ktiles) issue(it + 1, buf ^ 1);

        h16 afr[2];
#pragma unroll
        for (int mf = 0; mf < 2; ++mf) {
            const _Float16* p0 = &lA[buf][(wm * 32 + mf * 16 + ln) * 40 + g * 8];
            afr[mf] = frag_from2(p0, p0 + 16);
        }
#pragma unroll
        for (int nf = 0; nf < 4; ++nf) {
            const _Float16* p = &lBt[buf][(wn * 64 + nf * 16 + ln) * 40 + g * 8];
            h16 bfr = frag_from2(p, p + 16);
#pragma unroll
            for (int mf = 0; mf < 2; ++mf)
                acc[mf][nf] = __builtin_amdgcn_wmma_f32_16x16x32_f16(
                    false, afr[mf], false, bfr, (short)0, acc[mf][nf], false, false);
        }
        __syncthreads();         // done reading buf before tile it+2 overwrites it
    }

#pragma unroll
    for (int mf = 0; mf < 2; ++mf)
#pragma unroll
        for (int nf = 0; nf < 4; ++nf) {
            const int col = n0 + wn * 64 + nf * 16 + ln;
            const float bv = bias[col];
#pragma unroll
            for (int r = 0; r < 8; ++r) {
                const int row = m0 + wm * 32 + mf * 16 + r + 8 * g;
                float v = acc[mf][nf][r] + bv;
                if (RESID) v += resid[(size_t)row * N + col];
                if (RELU)  v = fmaxf(v, 0.f);
                if (OUTMODE == 0) {
                    C32[(size_t)row * N + col] = v;
                } else if (OUTMODE == 1) {
                    C16[(size_t)row * N + col] = (_Float16)v;
                } else {         // V-transposed: row=(b,s), col=(h,hd) -> [b,h][hd][s]
                    const int b = row >> 11, s = row & (S - 1);
                    const int h = col >> 6,  hd = col & (HD - 1);
                    C16[(((size_t)b * H + h) * HD + hd) * S + s] = (_Float16)v;
                }
            }
        }
}

// ---------------- Flash attention ----------------
// q,k: f16 (B,S,H,HD); vt: f16 [b,h][hd][s]; out: f16 (B,S,H,HD)
// grid (S/128, H, B); 256 threads = 8 waves, 16 query rows/wave; key tile 64; double-buffered async.
__global__ __launch_bounds__(256)
void k_flash_attn(const _Float16* __restrict__ q, const _Float16* __restrict__ k,
                  const _Float16* __restrict__ vt, _Float16* __restrict__ out) {
    __shared__ _Float16 lK[2][64 * 72];          // [key][hd]  stride 72
    __shared__ _Float16 lVt[2][64 * 72];         // [hd][key]  stride 72
    __shared__ _Float16 lP[8 * 16 * 72];         // per-wave P, [row][key] stride 72

    const int tid = threadIdx.x;
    const int l   = tid & 31;
    const int w   = tid >> 5;
    const int g   = l >> 4, ln = l & 15;
    const int qb  = blockIdx.x, h = blockIdx.y, b = blockIdx.z;

    const size_t base_bh = (size_t)b * S * D + (size_t)h * HD;          // q/k/out
    const _Float16* vbase = vt + ((size_t)b * H + h) * HD * S;          // [hd][s]

    auto issue = [&](int kt, int buf) {
#pragma unroll
        for (int j = 0; j < 2; ++j) {           // K tile 64 keys x 64 hd (row-major copy)
            int chunk = tid + 256 * j;
            int key = chunk >> 3, co = (chunk & 7) * 8;
            async_b128(&k[base_bh + (size_t)(kt * 64 + key) * D + co], &lK[buf][key * 72 + co]);
        }
#pragma unroll
        for (int j = 0; j < 2; ++j) {           // V tile already transposed in global
            int chunk = tid + 256 * j;
            int hd = chunk >> 3, kco = (chunk & 7) * 8;
            async_b128(&vbase[(size_t)hd * S + kt * 64 + kco], &lVt[buf][hd * 72 + kco]);
        }
    };

    // Q fragments for this wave's 16 rows (resident whole kernel)
    const int qrow = qb * 128 + w * 16 + ln;
    const _Float16* qp = q + base_bh + (size_t)qrow * D;
    h16 qa[2];
#pragma unroll
    for (int ks = 0; ks < 2; ++ks) {
        const _Float16* p0 = qp + ks * 32 + g * 8;
        qa[ks] = frag_from2(p0, p0 + 16);
    }

    f8 o[4];
#pragma unroll
    for (int nf = 0; nf < 4; ++nf)
#pragma unroll
        for (int r = 0; r < 8; ++r) o[nf][r] = 0.f;
    float mi[8], li[8];
#pragma unroll
    for (int r = 0; r < 8; ++r) { mi[r] = -INFINITY; li[r] = 0.f; }

    _Float16* lpw = &lP[w * 16 * 72];

    issue(0, 0);
    for (int kt = 0; kt < S / 64; ++kt) {
        const int buf = kt & 1;
        wait_async_all();
        __syncthreads();
        if (kt + 1 < S / 64) issue(kt + 1, buf ^ 1);

        // scores = Q @ K^T (16 x 64 per wave), scaled by 1/sqrt(HD)
        f8 s[4];
#pragma unroll
        for (int nf = 0; nf < 4; ++nf) {
            f8 a;
#pragma unroll
            for (int r = 0; r < 8; ++r) a[r] = 0.f;
#pragma unroll
            for (int ks = 0; ks < 2; ++ks) {
                const _Float16* p = &lK[buf][(nf * 16 + ln) * 72 + ks * 32 + g * 8];
                h16 bf = frag_from2(p, p + 16);
                a = __builtin_amdgcn_wmma_f32_16x16x32_f16(false, qa[ks], false, bf,
                                                           (short)0, a, false, false);
            }
#pragma unroll
            for (int r = 0; r < 8; ++r) a[r] *= 0.125f;
            s[nf] = a;
        }

        // online softmax
#pragma unroll
        for (int r = 0; r < 8; ++r) {
            float mx = s[0][r];
#pragma unroll
            for (int nf = 1; nf < 4; ++nf) mx = fmaxf(mx, s[nf][r]);
            mx = red_max16(mx);
            const float mn = fmaxf(mi[r], mx);
            const float sc = __expf(mi[r] - mn);
            mi[r] = mn;
            float rs = 0.f;
#pragma unroll
            for (int nf = 0; nf < 4; ++nf) {
                float p = __expf(s[nf][r] - mn);
                s[nf][r] = p;
                rs += p;
            }
            rs = red_sum16(rs);
            li[r] = li[r] * sc + rs;
#pragma unroll
            for (int nf = 0; nf < 4; ++nf) o[nf][r] *= sc;
        }

        // P (C-layout f32) -> per-wave LDS f16 (A-layout source)
#pragma unroll
        for (int nf = 0; nf < 4; ++nf)
#pragma unroll
            for (int r = 0; r < 8; ++r)
                lpw[(r + 8 * g) * 72 + nf * 16 + ln] = (_Float16)s[nf][r];

        // O += P @ V
#pragma unroll
        for (int ks = 0; ks < 2; ++ks) {
            const _Float16* pp = &lpw[ln * 72 + ks * 32 + g * 8];
            h16 pa = frag_from2(pp, pp + 16);
#pragma unroll
            for (int nf = 0; nf < 4; ++nf) {
                const _Float16* p = &lVt[buf][(nf * 16 + ln) * 72 + ks * 32 + g * 8];
                h16 bv = frag_from2(p, p + 16);
                o[nf] = __builtin_amdgcn_wmma_f32_16x16x32_f16(false, pa, false, bv,
                                                               (short)0, o[nf], false, false);
            }
        }
        __syncthreads();
    }

    // normalize + store
#pragma unroll
    for (int nf = 0; nf < 4; ++nf)
#pragma unroll
        for (int r = 0; r < 8; ++r) {
            const int srow = qb * 128 + w * 16 + r + 8 * g;
            out[base_bh + (size_t)srow * D + nf * 16 + ln] = (_Float16)(o[nf][r] / li[r]);
        }
}

// ---------------- driver ----------------
extern "C" void kernel_launch(void* const* d_in, const int* in_sizes, int n_in,
                              void* d_out, int out_size, void* d_ws, size_t ws_size,
                              hipStream_t stream) {
    const float* x    = (const float*)d_in[0];
    const float* wq   = (const float*)d_in[1];
    const float* bq   = (const float*)d_in[2];
    const float* wk   = (const float*)d_in[3];
    const float* bk   = (const float*)d_in[4];
    const float* wv   = (const float*)d_in[5];
    const float* bv   = (const float*)d_in[6];
    const float* wo   = (const float*)d_in[7];
    const float* bo   = (const float*)d_in[8];
    const float* w1   = (const float*)d_in[9];
    const float* b1   = (const float*)d_in[10];
    const float* w2   = (const float*)d_in[11];
    const float* b2   = (const float*)d_in[12];
    const float* ln1a = (const float*)d_in[13];
    const float* ln1b = (const float*)d_in[14];
    const float* ln2a = (const float*)d_in[15];
    const float* ln2b = (const float*)d_in[16];
    float* outp = (float*)d_out;

    char* p = (char*)d_ws;
    auto takeh = [&](size_t n) { _Float16* r = (_Float16*)p; p += n * sizeof(_Float16); return r; };
    auto takef = [&](size_t n) { float* r = (float*)p;       p += n * sizeof(float);    return r; };
    _Float16* wqp = takeh((size_t)D * D);      // packed weights (f16, tile-major)
    _Float16* wkp = takeh((size_t)D * D);
    _Float16* wvp = takeh((size_t)D * D);
    _Float16* wop = takeh((size_t)D * D);
    _Float16* w1p = takeh((size_t)D * DFF);
    _Float16* w2p = takeh((size_t)DFF * D);
    _Float16* h1  = takeh((size_t)M * D);      // LN1 out
    _Float16* q16 = takeh((size_t)M * D);
    _Float16* k16 = takeh((size_t)M * D);
    _Float16* vtr = takeh((size_t)M * D);      // V, transposed [b,h][hd][s]
    _Float16* at16= takeh((size_t)M * D);      // attention out
    float*    x1  = takef((size_t)M * D);      // residual-1 (fp32)
    _Float16* h2  = takeh((size_t)M * D);      // LN2 out
    _Float16* f1  = takeh((size_t)M * DFF);    // relu(h2@w1+b1)

    // 1) repack weights f32 -> f16 tile-major [n/128][k/32][n128][k32]
    k_pack_wt<<<2048, 256, 0, stream>>>(wq, wqp, D, D);
    k_pack_wt<<<2048, 256, 0, stream>>>(wk, wkp, D, D);
    k_pack_wt<<<2048, 256, 0, stream>>>(wv, wvp, D, D);
    k_pack_wt<<<2048, 256, 0, stream>>>(wo, wop, D, D);
    k_pack_wt<<<4096, 256, 0, stream>>>(w1, w1p, D, DFF);
    k_pack_wt<<<4096, 256, 0, stream>>>(w2, w2p, DFF, D);

    // 2) LN1
    k_layernorm<<<M, 256, 0, stream>>>(x, ln1a, ln1b, h1);

    // 3) Q,K,V projections
    dim3 gqkv(D / 128, M / 128);
    k_gemm_wmma<false, false, 1><<<gqkv, 256, 0, stream>>>(h1, wqp, bq, nullptr, nullptr, q16, D, D);
    k_gemm_wmma<false, false, 1><<<gqkv, 256, 0, stream>>>(h1, wkp, bk, nullptr, nullptr, k16, D, D);
    k_gemm_wmma<false, false, 2><<<gqkv, 256, 0, stream>>>(h1, wvp, bv, nullptr, nullptr, vtr, D, D);

    // 4) flash attention
    dim3 gfa(S / 128, H, Bb);
    k_flash_attn<<<gfa, 256, 0, stream>>>(q16, k16, vtr, at16);

    // 5) output projection + residual (fp32)
    k_gemm_wmma<false, true, 0><<<gqkv, 256, 0, stream>>>(at16, wop, bo, x, x1, nullptr, D, D);

    // 6) LN2
    k_layernorm<<<M, 256, 0, stream>>>(x1, ln2a, ln2b, h2);

    // 7) FFN1 + ReLU
    dim3 gff1(DFF / 128, M / 128);
    k_gemm_wmma<true, false, 1><<<gff1, 256, 0, stream>>>(h2, w1p, b1, nullptr, nullptr, f1, DFF, D);

    // 8) FFN2 + residual -> d_out (fp32)
    k_gemm_wmma<false, true, 0><<<gqkv, 256, 0, stream>>>(f1, w2p, b2, x1, outp, nullptr, D, DFF);
}
